// GAT_Default_7567732376257
// MI455X (gfx1250) — matrix operand order repaired and checked by reading, not compiled, not run on hardware
//
#include <hip/hip_runtime.h>
#include <hip/hip_bf16.h>
#include <math.h>

typedef __attribute__((ext_vector_type(16))) _Float16 v16h;
typedef __attribute__((ext_vector_type(8)))  float    v8f;

#define NEG_SLOPE 0.2f

// ---------------------------------------------------------------------------
// helpers
// ---------------------------------------------------------------------------
__device__ __forceinline__ void edge_nodes(long long e, const long long* __restrict__ ei,
                                           long long E, int& s, int& d) {
  if (e < E) {
    s = (int)ei[e];        // edge_index[0][e]
    d = (int)ei[E + e];    // edge_index[1][e]
  } else {                 // appended self loop
    int n = (int)(e - E);
    s = n; d = n;
  }
}

// float atomic max via ordered-int trick (works for mixed-sign, init = -inf)
__device__ __forceinline__ void atomicMaxF(float* addr, float val) {
  if (val >= 0.0f) atomicMax((int*)addr, __float_as_int(val));
  else             atomicMin((unsigned int*)addr, __float_as_uint(val));
}

__global__ void fill_f32(float* __restrict__ p, float v, long long n) {
  long long i = (long long)blockIdx.x * blockDim.x + threadIdx.x;
  if (i < n) p[i] = v;
}

// decode int64 edge_index (+ implicit self loops) into int32 src/dst once
__global__ void prep_edges(const long long* __restrict__ ei, long long E, long long ET,
                           int* __restrict__ s32, int* __restrict__ d32) {
  long long e = (long long)blockIdx.x * blockDim.x + threadIdx.x;
  if (e >= ET) return;
  int s, d; edge_nodes(e, ei, E, s, d);
  s32[e] = s; d32[e] = d;
}

// ---------------------------------------------------------------------------
// Pack W[128,NC] (f32, row major) into f16 per-lane WMMA B layout:
//   Wp[((ks*(NC/16) + ntile)*32 + lane)*16 + i]  == B element K=ks*32+kk, N
//   where lanes 0-15 hold kk=0..15 (i=kk) for col N=ntile*16+lane,
//         lanes 16-31 hold kk=16..31.
// In the GEMM each lane then loads one contiguous 32B v16h (coalesced).
// ---------------------------------------------------------------------------
__global__ void pack_B(const float* __restrict__ W, _Float16* __restrict__ Wp, int NC) {
  int idx = blockIdx.x * blockDim.x + threadIdx.x;    // over 128*NC
  if (idx >= 128 * NC) return;
  int k = idx / NC, n = idx % NC;
  int ks = k >> 5, krem = k & 31;
  int tile = n >> 4, nl = n & 15;
  int lane = (krem < 16) ? nl : (16 + nl);
  int i = krem & 15;
  Wp[(((ks * (NC >> 4) + tile) * 32) + lane) * 16 + i] = (_Float16)W[idx];
}

// ---------------------------------------------------------------------------
// WMMA GEMM:  Out[M,NC] = X[M,128] @ W[128,NC]   (f16 in, f32 accumulate)
// one block = 16-row M tile; one wave = 16-col N tile; K=128 in 4 WMMA steps.
// A staged in LDS with padded stride (bank-conflict free); B pre-packed;
// D staged in LDS for coalesced float4 writeback.
// ---------------------------------------------------------------------------
template<int NC, bool XHALF>
__global__ __launch_bounds__(NC * 2) void gemm_wmma(
    const float* __restrict__ Xf, const _Float16* __restrict__ Xh,
    const _Float16* __restrict__ Wp, float* __restrict__ Out, int M) {
  constexpr int LDA = 136;                 // 272B row stride: 16B aligned, bank-skewed
  __shared__ _Float16 sA[16 * LDA];
  __shared__ float    sD[16 * NC];
  const int tid  = threadIdx.x;
  const int row0 = blockIdx.x * 16;

  for (int i = tid; i < 16 * 128; i += NC * 2) {
    int r  = i >> 7;
    int c  = i & 127;
    int rg = row0 + r;
    _Float16 v = (_Float16)0.0f;
    if (rg < M)
      v = XHALF ? Xh[(long long)rg * 128 + c] : (_Float16)Xf[(long long)rg * 128 + c];
    sA[r * LDA + c] = v;
  }
  __syncthreads();

  const int  wave = tid >> 5;
  const int  lane = tid & 31;
  const int  lrow = lane & 15;
  const bool hi   = lane >= 16;
  const v16h* WpV = (const v16h*)Wp;

  v8f acc = {};
#pragma unroll
  for (int ks = 0; ks < 4; ++ks) {
    v16h a;
    // A 16x32 f16, contiguous k as in packed B: lanes 0-15 row M=lane, K=0..15; lanes 16-31
    // same rows, K=16..31 -> two ds_load_b128 per lane
    const _Float16* ap = &sA[lrow * LDA + ks * 32 + (hi ? 16 : 0)];
#pragma unroll
    for (int i = 0; i < 8; ++i) { a[i] = ap[i]; a[8 + i] = ap[8 + i]; }
    v16h b = WpV[(ks * (NC >> 4) + wave) * 32 + lane];   // contiguous, coalesced
    acc = __builtin_amdgcn_wmma_f32_16x16x32_f16(false, a, false, b,
                                                 (short)0, acc, false, false);
  }

  // D layout: VGPR r -> row r (lanes 0-15) / row 8+r (lanes 16-31), col = lane%16
  const int drow0 = hi ? 8 : 0;
  const int col   = wave * 16 + lrow;
#pragma unroll
  for (int r = 0; r < 8; ++r) sD[(drow0 + r) * NC + col] = acc[r];
  __syncthreads();

  // coalesced writeback, float4 per lane
  for (int i = tid; i < 16 * NC / 4; i += NC * 2) {
    int r  = (i * 4) / NC;
    int c  = (i * 4) % NC;
    int rg = row0 + r;
    if (rg < M)
      *(float4*)&Out[(long long)rg * NC + c] = *(const float4*)&sD[r * NC + c];
  }
}

// ---------------------------------------------------------------------------
// per-(node,head) attention coefficients: a_src/a_dst = sum_c h[n,h,c]*att[h,c]
// ---------------------------------------------------------------------------
__global__ void attn_coef(const float* __restrict__ h,
                          const float* __restrict__ att_src,
                          const float* __restrict__ att_dst,
                          float* __restrict__ asrc, float* __restrict__ adst,
                          int n, int H, int C) {
  int t = blockIdx.x * blockDim.x + threadIdx.x;   // t = node*H + head
  if (t >= n * H) return;
  int head = t % H;
  const float* hp = &h[(long long)t * C];
  float s = 0.f, d = 0.f;
  for (int c = 0; c < C; ++c) {
    float v = hp[c];
    s += v * att_src[head * C + c];
    d += v * att_dst[head * C + c];
  }
  asrc[t] = s;
  adst[t] = d;
}

template<int H>
__device__ __forceinline__ float edge_logit(const float* asrc, const float* adst,
                                            int s, int d, int head) {
  float v = asrc[(long long)s * H + head] + adst[(long long)d * H + head];
  return v > 0.f ? v : NEG_SLOPE * v;
}

template<int H>
__global__ void edge_max(const int* __restrict__ s32, const int* __restrict__ d32,
                         long long ET,
                         const float* __restrict__ asrc, const float* __restrict__ adst,
                         float* __restrict__ m) {
  long long t = (long long)blockIdx.x * blockDim.x + threadIdx.x;
  if (t >= ET * H) return;
  long long e = t / H;
  int head = (int)(t % H);
  int s = s32[e], d = d32[e];
  atomicMaxF(&m[(long long)d * H + head], edge_logit<H>(asrc, adst, s, d, head));
}

template<int H>
__global__ void edge_sum(const int* __restrict__ s32, const int* __restrict__ d32,
                         long long ET,
                         const float* __restrict__ asrc, const float* __restrict__ adst,
                         const float* __restrict__ m, float* __restrict__ denom) {
  long long t = (long long)blockIdx.x * blockDim.x + threadIdx.x;
  if (t >= ET * H) return;
  long long e = t / H;
  int head = (int)(t % H);
  int s = s32[e], d = d32[e];
  float v = edge_logit<H>(asrc, adst, s, d, head);
  atomicAdd(&denom[(long long)d * H + head], __expf(v - m[(long long)d * H + head]));
}

// one wave per edge; lane = channel (C == 32 == wave32 lanes for both layers)
template<int H>
__global__ void edge_agg(const int* __restrict__ s32, const int* __restrict__ d32,
                         long long ET,
                         const float* __restrict__ asrc, const float* __restrict__ adst,
                         const float* __restrict__ m, const float* __restrict__ denom,
                         const float* __restrict__ h, float* __restrict__ out) {
  long long wid = ((long long)blockIdx.x * blockDim.x + threadIdx.x) >> 5;
  int lane = threadIdx.x & 31;
  if (wid >= ET) return;
  int s = s32[wid], d = d32[wid];
#pragma unroll
  for (int head = 0; head < H; ++head) {
    float v = edge_logit<H>(asrc, adst, s, d, head);
    float alpha = __expf(v - m[(long long)d * H + head]) /
                  (denom[(long long)d * H + head] + 1e-16f);
    atomicAdd(&out[((long long)d * H + head) * 32 + lane],
              h[((long long)s * H + head) * 32 + lane] * alpha);
  }
}

// x2 = f16(relu(agg1 + b1)), feeds layer-2 WMMA GEMM
__global__ void post1(const float* __restrict__ agg, const float* __restrict__ b1,
                      _Float16* __restrict__ x2, long long n) {
  long long i = (long long)blockIdx.x * blockDim.x + threadIdx.x;
  if (i < n) {
    float v = agg[i] + b1[i & 127];
    x2[i] = (_Float16)(v > 0.f ? v : 0.f);
  }
}

__global__ void bias_add(float* __restrict__ out, const float* __restrict__ b,
                         long long n, int C) {
  long long i = (long long)blockIdx.x * blockDim.x + threadIdx.x;
  if (i < n) out[i] += b[i % C];
}

// ---------------------------------------------------------------------------
extern "C" void kernel_launch(void* const* d_in, const int* in_sizes, int n_in,
                              void* d_out, int out_size, void* d_ws, size_t ws_size,
                              hipStream_t stream) {
  const float*     x     = (const float*)d_in[0];
  const long long* ei    = (const long long*)d_in[1];   // int64 [2,E]
  const float*     W1    = (const float*)d_in[2];
  const float*     at_s1 = (const float*)d_in[3];
  const float*     at_d1 = (const float*)d_in[4];
  const float*     b1    = (const float*)d_in[5];
  const float*     W2    = (const float*)d_in[6];
  const float*     at_s2 = (const float*)d_in[7];
  const float*     at_d2 = (const float*)d_in[8];
  const float*     b2    = (const float*)d_in[9];

  const int       N  = in_sizes[0] / 128;     // 100000
  const long long E  = in_sizes[1] / 2;       // 1600000
  const long long ET = E + N;                 // + self loops

  // workspace carve-out (256B aligned)
  char*  base = (char*)d_ws;
  size_t off  = 0;
  auto take = [&](size_t bytes) -> void* {
    void* p = base + off;
    off += (bytes + 255) & ~(size_t)255;
    return p;
  };
  _Float16* W1p  = (_Float16*)take(128 * 128 * sizeof(_Float16));
  _Float16* W2p  = (_Float16*)take(128 * 32 * sizeof(_Float16));
  float*    h1   = (float*)take((size_t)N * 128 * 4);
  float*    as1  = (float*)take((size_t)N * 4 * 4);
  float*    ad1  = (float*)take((size_t)N * 4 * 4);
  float*    m1   = (float*)take((size_t)N * 4 * 4);
  float*    dn1  = (float*)take((size_t)N * 4 * 4);
  float*    agg1 = (float*)take((size_t)N * 128 * 4);
  _Float16* x2h  = (_Float16*)take((size_t)N * 128 * 2);
  float*    h2   = (float*)take((size_t)N * 32 * 4);
  float*    as2  = (float*)take((size_t)N * 4);
  float*    ad2  = (float*)take((size_t)N * 4);
  float*    m2   = (float*)take((size_t)N * 4);
  float*    dn2  = (float*)take((size_t)N * 4);
  int*      s32  = (int*)take((size_t)ET * 4);
  int*      d32  = (int*)take((size_t)ET * 4);
  float*    outp = (float*)d_out;

  const int T = 256;
  auto nb = [&](long long n) { return (unsigned)((n + T - 1) / T); };

  // re-init every call (graph-replay safe)
  fill_f32<<<nb((long long)N * 4), T, 0, stream>>>(m1, -INFINITY, (long long)N * 4);
  fill_f32<<<nb((long long)N * 4), T, 0, stream>>>(dn1, 0.0f, (long long)N * 4);
  fill_f32<<<nb((long long)N * 128), T, 0, stream>>>(agg1, 0.0f, (long long)N * 128);
  fill_f32<<<nb(N), T, 0, stream>>>(m2, -INFINITY, N);
  fill_f32<<<nb(N), T, 0, stream>>>(dn2, 0.0f, N);
  fill_f32<<<nb((long long)N * 32), T, 0, stream>>>(outp, 0.0f, (long long)N * 32);

  prep_edges<<<nb(ET), T, 0, stream>>>(ei, E, ET, s32, d32);
  pack_B<<<nb(128 * 128), T, 0, stream>>>(W1, W1p, 128);
  pack_B<<<nb(128 * 32), T, 0, stream>>>(W2, W2p, 32);

  const unsigned mtiles = (unsigned)((N + 15) / 16);

  // ---- layer 1 ----
  gemm_wmma<128, false><<<mtiles, 256, 0, stream>>>(x, nullptr, W1p, h1, N);
  attn_coef<<<nb((long long)N * 4), T, 0, stream>>>(h1, at_s1, at_d1, as1, ad1, N, 4, 32);
  edge_max<4><<<nb(ET * 4), T, 0, stream>>>(s32, d32, ET, as1, ad1, m1);
  edge_sum<4><<<nb(ET * 4), T, 0, stream>>>(s32, d32, ET, as1, ad1, m1, dn1);
  edge_agg<4><<<nb(ET * 32), T, 0, stream>>>(s32, d32, ET, as1, ad1, m1, dn1, h1, agg1);
  post1<<<nb((long long)N * 128), T, 0, stream>>>(agg1, b1, x2h, (long long)N * 128);

  // ---- layer 2 ----
  gemm_wmma<32, true><<<mtiles, 64, 0, stream>>>(nullptr, x2h, W2p, h2, N);
  attn_coef<<<nb(N), T, 0, stream>>>(h2, at_s2, at_d2, as2, ad2, N, 1, 32);
  edge_max<1><<<nb(ET), T, 0, stream>>>(s32, d32, ET, as2, ad2, m2);
  edge_sum<1><<<nb(ET), T, 0, stream>>>(s32, d32, ET, as2, ad2, m2, dn2);
  edge_agg<1><<<nb(ET * 32), T, 0, stream>>>(s32, d32, ET, as2, ad2, m2, dn2, h2, outp);
  bias_add<<<nb((long long)N * 32), T, 0, stream>>>(outp, b2, (long long)N * 32, 32);
}
